// Llama_26439818674223
// MI455X (gfx1250) — matrix operand order repaired and checked by reading, not compiled, not run on hardware
//
#include <hip/hip_runtime.h>

typedef __bf16 bf16;
typedef __attribute__((ext_vector_type(16))) __bf16 bf16x16;
typedef __attribute__((ext_vector_type(8))) float f32x8;

union FragB { uint4 u[2]; bf16x16 v; };

// ---------------------------------------------------------------------------
// embed: x[i][d] = token_emb[tokens[i]][d]   (4096 x 1024 f32)
// ---------------------------------------------------------------------------
__global__ __launch_bounds__(256)
void embed_kernel(const int* __restrict__ tok, const float* __restrict__ emb,
                  float* __restrict__ x)
{
    int idx = blockIdx.x * 256 + threadIdx.x;      // exactly 4096*1024 threads
    int d  = idx & 1023;
    int rw = idx >> 10;
    x[idx] = emb[(size_t)tok[rw] * 1024 + d];
}

// ---------------------------------------------------------------------------
// weight convert + transpose + pad:  Wt[nn][kk] = (bf16) W[kk][nn], zero pad
// ---------------------------------------------------------------------------
__global__ __launch_bounds__(256)
void wconv_kernel(const float* __restrict__ W, bf16* __restrict__ Wt,
                  int K, int N, int Kp, long total)
{
    long idx = (long)blockIdx.x * 256 + threadIdx.x;
    if (idx >= total) return;
    int kk = (int)(idx % Kp);
    long nn = idx / Kp;
    float v = (kk < K && nn < N) ? W[(size_t)kk * N + nn] : 0.f;
    Wt[idx] = (bf16)v;
}

// ---------------------------------------------------------------------------
// rmsnorm over dim=1024, write bf16
// ---------------------------------------------------------------------------
__global__ __launch_bounds__(256)
void rmsnorm_kernel(const float* __restrict__ x, const float* __restrict__ w,
                    bf16* __restrict__ out)
{
    __shared__ float red[256];
    int rowb = blockIdx.x;
    const float* xr = x + (size_t)rowb * 1024;
    float vals[4];
    float s = 0.f;
#pragma unroll
    for (int i = 0; i < 4; ++i) {
        vals[i] = xr[threadIdx.x + 256 * i];
        s += vals[i] * vals[i];
    }
    red[threadIdx.x] = s;
    __syncthreads();
    for (int off = 128; off > 0; off >>= 1) {
        if (threadIdx.x < off) red[threadIdx.x] += red[threadIdx.x + off];
        __syncthreads();
    }
    float rinv = rsqrtf(red[0] * (1.f / 1024.f) + 1.1920928955078125e-7f);
#pragma unroll
    for (int i = 0; i < 4; ++i) {
        int d = threadIdx.x + 256 * i;
        out[(size_t)rowb * 1024 + d] = (bf16)(vals[i] * rinv * w[d]);
    }
}

// ---------------------------------------------------------------------------
// GEMM: C[M,N] = A[M,Kp](bf16) * Bt[Npad,Kp](bf16)^T  (+bias)(+resid)
// 8 waves / block; each wave -> 32x64 strip (2 row-tiles x 4 col-tiles).
// Software-pipelined: K unrolled by 64 with two fragment buffers so next-step
// loads are issued before current-step WMMAs (partial s_wait_loadcnt).
// A frag: lane(l%16=row, h=l/16): K = h*8+[0..7] , 16+h*8+[0..7]  -> 2x b128
// B frag: lane(l%16=col, h):      K = h*8+[0..7] , 16+h*8+[0..7]  -> 2x b128
// Requires Kp % 64 == 0 (all Kp here are 1024 or 2752).
// ---------------------------------------------------------------------------
struct KStep { FragB a[2]; FragB b[4]; };

__device__ __forceinline__ void load_kstep(KStep& ks,
    const bf16* a0, const bf16* a1,
    const bf16* b0, const bf16* b1, const bf16* b2, const bf16* b3,
    int k, int hl)
{
    ks.a[0].u[0] = *(const uint4*)(a0 + k + hl * 8);
    ks.a[0].u[1] = *(const uint4*)(a0 + k + 16 + hl * 8);
    ks.a[1].u[0] = *(const uint4*)(a1 + k + hl * 8);
    ks.a[1].u[1] = *(const uint4*)(a1 + k + 16 + hl * 8);
    ks.b[0].u[0] = *(const uint4*)(b0 + k);
    ks.b[0].u[1] = *(const uint4*)(b0 + k + 16);
    ks.b[1].u[0] = *(const uint4*)(b1 + k);
    ks.b[1].u[1] = *(const uint4*)(b1 + k + 16);
    ks.b[2].u[0] = *(const uint4*)(b2 + k);
    ks.b[2].u[1] = *(const uint4*)(b2 + k + 16);
    ks.b[3].u[0] = *(const uint4*)(b3 + k);
    ks.b[3].u[1] = *(const uint4*)(b3 + k + 16);
}

__device__ __forceinline__ void mma_step(f32x8 acc[2][4], const KStep& ks)
{
#pragma unroll
    for (int r = 0; r < 2; ++r)
#pragma unroll
        for (int t = 0; t < 4; ++t)
            acc[r][t] = __builtin_amdgcn_wmma_f32_16x16x32_bf16(
                false, ks.a[r].v, false, ks.b[t].v, (short)0, acc[r][t], false, false);
}

__global__ __launch_bounds__(256)
void gemm_bf16_kernel(const bf16* __restrict__ A, const bf16* __restrict__ Bt,
                      float* __restrict__ C, const float* __restrict__ bias,
                      const float* __restrict__ resid, int M, int N, int Kp)
{
    int lane = threadIdx.x & 31;
    int wave = threadIdx.x >> 5;
    int m0 = blockIdx.y * 256 + wave * 32;
    int n0 = blockIdx.x * 64;
    int rl = lane & 15, hl = lane >> 4;
    if (m0 >= M) return;   // wave-uniform

    f32x8 acc[2][4];
#pragma unroll
    for (int r = 0; r < 2; ++r)
#pragma unroll
        for (int t = 0; t < 4; ++t)
#pragma unroll
            for (int v = 0; v < 8; ++v) acc[r][t][v] = 0.f;

    const bf16* a0 = A + (size_t)(m0 + rl) * Kp;
    const bf16* a1 = a0 + (size_t)16 * Kp;
    const bf16* b0 = Bt + (size_t)(n0 + rl) * Kp + hl * 8;
    const bf16* b1 = b0 + (size_t)16 * Kp;
    const bf16* b2 = b0 + (size_t)32 * Kp;
    const bf16* b3 = b0 + (size_t)48 * Kp;

    KStep s0, s1;
    load_kstep(s0, a0, a1, b0, b1, b2, b3, 0, hl);
    for (int k0 = 0; k0 < Kp - 64; k0 += 64) {
        load_kstep(s1, a0, a1, b0, b1, b2, b3, k0 + 32, hl);
        mma_step(acc, s0);
        load_kstep(s0, a0, a1, b0, b1, b2, b3, k0 + 64, hl);
        mma_step(acc, s1);
    }
    load_kstep(s1, a0, a1, b0, b1, b2, b3, Kp - 32, hl);
    mma_step(acc, s0);
    mma_step(acc, s1);

    // C layout: lane holds (rows hl*8+v, col l%16) per 16x16 tile
#pragma unroll
    for (int r = 0; r < 2; ++r) {
#pragma unroll
        for (int t = 0; t < 4; ++t) {
            int c = n0 + t * 16 + rl;
            if (c < N) {
#pragma unroll
                for (int v = 0; v < 8; ++v) {
                    int rr = m0 + r * 16 + hl * 8 + v;
                    float val = acc[r][t][v];
                    if (bias)  val += bias[c];
                    if (resid) val += resid[(size_t)rr * N + c];
                    C[(size_t)rr * N + c] = val;
                }
            }
        }
    }
}

// ---------------------------------------------------------------------------
// RoPE + scale + bf16 split of qkv[4096,3072] into
//   q[bh][n][64], k[bh][n][64], vT[bh][64][n]   (bh = b*16+h, n = 2048)
// ---------------------------------------------------------------------------
__global__ __launch_bounds__(256)
void rope_kernel(const float* __restrict__ qkv, bf16* __restrict__ qo,
                 bf16* __restrict__ ko, bf16* __restrict__ vT)
{
    int idx = blockIdx.x * 256 + threadIdx.x;      // 32*2048*64 threads
    int d  = idx & 63;
    int n  = (idx >> 6) & 2047;
    int bh = idx >> 17;
    int b = bh >> 4, h = bh & 15;
    size_t rowbase = ((size_t)(b * 2048 + n)) * 3072 + h * 64 + d;
    float qv = qkv[rowbase];
    float kv = qkv[rowbase + 1024];
    float vv = qkv[rowbase + 2048];
    int j = d & 31;
    float inv = __expf(-(float)j * (9.210340371976184f / 32.f)); // 10000^(-j/32)
    float f = (float)n * inv;
    float sn, cs;
    __sincosf(f, &sn, &cs);
    float qr, kr;
    if (d < 32) {
        qr = qv * cs - qkv[rowbase + 32] * sn;
        kr = kv * cs - qkv[rowbase + 32 + 1024] * sn;
    } else {
        qr = qv * cs + qkv[rowbase - 32] * sn;
        kr = kv * cs + qkv[rowbase - 32 + 1024] * sn;
    }
    qr *= 0.125f; // dh^-0.5
    size_t o = ((size_t)bh * 2048 + n) * 64 + d;
    qo[o] = (bf16)qr;
    ko[o] = (bf16)kr;
    vT[((size_t)bh * 64 + d) * 2048 + n] = (bf16)vv;
}

// ---------------------------------------------------------------------------
// Flash attention: 1 wave per (bh, 16-query tile); causal; out bf16 [4096,1024]
// ---------------------------------------------------------------------------
__global__ __launch_bounds__(32)
void attn_kernel(const bf16* __restrict__ q, const bf16* __restrict__ k,
                 const bf16* __restrict__ vT, bf16* __restrict__ out)
{
    __shared__ bf16 Pl[16][32];
    int lane = threadIdx.x & 31;
    int row = lane & 15, half = lane >> 4;
    int qt = blockIdx.x, bh = blockIdx.y;
    int b = bh >> 4, h = bh & 15;
    int q0 = qt * 16;

    FragB qf[2];
    const bf16* qrow = q + ((size_t)bh * 2048 + q0 + row) * 64;
#pragma unroll
    for (int c = 0; c < 2; ++c) {
        qf[c].u[0] = *(const uint4*)(qrow + c * 32 + half * 8);
        qf[c].u[1] = *(const uint4*)(qrow + c * 32 + 16 + half * 8);
    }

    f32x8 oacc[4];
    float m[8], s[8];
#pragma unroll
    for (int t = 0; t < 4; ++t)
#pragma unroll
        for (int v = 0; v < 8; ++v) oacc[t][v] = 0.f;
#pragma unroll
    for (int v = 0; v < 8; ++v) { m[v] = -3.0e38f; s[v] = 0.f; }

    for (int j = 0; j <= q0 + 15; j += 32) {          // uniform per wave
        f32x8 sacc[2];
#pragma unroll
        for (int t = 0; t < 2; ++t)
#pragma unroll
            for (int v = 0; v < 8; ++v) sacc[t][v] = 0.f;
#pragma unroll
        for (int t = 0; t < 2; ++t) {
            const bf16* krow = k + ((size_t)bh * 2048 + j + t * 16 + row) * 64;
#pragma unroll
            for (int c = 0; c < 2; ++c) {
                FragB kf;
                kf.u[0] = *(const uint4*)(krow + c * 32 + half * 8);
                kf.u[1] = *(const uint4*)(krow + c * 32 + 16 + half * 8);
                sacc[t] = __builtin_amdgcn_wmma_f32_16x16x32_bf16(
                    false, qf[c].v, false, kf.v, (short)0, sacc[t], false, false);
            }
        }
        // online softmax update (rows half*8+v, cols = row within tile)
#pragma unroll
        for (int v = 0; v < 8; ++v) {
            int r = half * 8 + v;
            int qg = q0 + r;
            float s0 = (j + row      > qg) ? -3.0e38f : sacc[0][v];
            float s1 = (j + 16 + row > qg) ? -3.0e38f : sacc[1][v];
            float mx = fmaxf(s0, s1);
#pragma unroll
            for (int o = 1; o < 16; o <<= 1) mx = fmaxf(mx, __shfl_xor(mx, o, 16));
            float mnew = fmaxf(m[v], mx);
            float alpha = __expf(m[v] - mnew);
            float p0 = __expf(s0 - mnew);
            float p1 = __expf(s1 - mnew);
            float rs = p0 + p1;
#pragma unroll
            for (int o = 1; o < 16; o <<= 1) rs += __shfl_xor(rs, o, 16);
            s[v] = s[v] * alpha + rs;
            m[v] = mnew;
#pragma unroll
            for (int t = 0; t < 4; ++t) oacc[t][v] *= alpha;
            Pl[r][row]      = (bf16)p0;
            Pl[r][16 + row] = (bf16)p1;
        }
        // Re-read P in A-fragment layout (DS ops are in-order within a wave)
        FragB pf;
        pf.u[0] = *(const uint4*)(&Pl[row][half * 8]);
        pf.u[1] = *(const uint4*)(&Pl[row][16 + half * 8]);
#pragma unroll
        for (int cv = 0; cv < 4; ++cv) {
            const bf16* vrow = vT + ((size_t)bh * 64 + cv * 16 + row) * 2048 + j + half * 8;
            FragB vf;
            vf.u[0] = *(const uint4*)(vrow);
            vf.u[1] = *(const uint4*)(vrow + 16);
            oacc[cv] = __builtin_amdgcn_wmma_f32_16x16x32_bf16(
                false, pf.v, false, vf.v, (short)0, oacc[cv], false, false);
        }
    }
#pragma unroll
    for (int v = 0; v < 8; ++v) {
        int r = half * 8 + v;
        float inv = 1.f / s[v];
        size_t orow = ((size_t)(b * 2048 + q0 + r)) * 1024 + h * 64;
#pragma unroll
        for (int cv = 0; cv < 4; ++cv)
            out[orow + cv * 16 + row] = (bf16)(oacc[cv][v] * inv);
    }
}

// ---------------------------------------------------------------------------
// GEGLU: g[:, :2730] = gelu_exact(u[:,2730:]) * u[:,:2730]; pad K to 2752, bf16
// ---------------------------------------------------------------------------
__global__ __launch_bounds__(256)
void geglu_kernel(const float* __restrict__ u, bf16* __restrict__ g)
{
    int idx = blockIdx.x * 256 + threadIdx.x;
    if (idx >= 4096 * 2752) return;
    int c = idx % 2752;
    int r = idx / 2752;
    float val = 0.f;
    if (c < 2730) {
        float u1 = u[(size_t)r * 5460 + c];
        float gt = u[(size_t)r * 5460 + 2730 + c];
        val = 0.5f * gt * (1.f + erff(gt * 0.70710678118654752f)) * u1;
    }
    g[idx] = (bf16)val;
}

// ---------------------------------------------------------------------------
extern "C" void kernel_launch(void* const* d_in, const int* in_sizes, int n_in,
                              void* d_out, int out_size, void* d_ws, size_t ws_size,
                              hipStream_t stream)
{
    const int*   tokens       = (const int*)  d_in[0];
    const float* token_emb    = (const float*)d_in[1];
    const float* attn_norm_w  = (const float*)d_in[2];
    const float* wqkv         = (const float*)d_in[3];
    const float* wo           = (const float*)d_in[4];
    const float* ff_norm_w    = (const float*)d_in[5];
    const float* ff_w1        = (const float*)d_in[6];
    const float* ff_b1        = (const float*)d_in[7];
    const float* ff_w2        = (const float*)d_in[8];
    const float* ff_b2        = (const float*)d_in[9];
    const float* final_norm_w = (const float*)d_in[10];
    const float* logits_w     = (const float*)d_in[11];
    const float* logits_b     = (const float*)d_in[12];
    float* out = (float*)d_out;

    const int M = 4096;
    char* wsp = (char*)d_ws;
    auto alloc = [&](size_t bytes) {
        void* p = wsp;
        wsp += (bytes + 255) & ~(size_t)255;
        return p;
    };
    float* x    = (float*)alloc((size_t)M * 1024 * 4);
    float* x2   = (float*)alloc((size_t)M * 1024 * 4);
    bf16*  h    = (bf16*) alloc((size_t)M * 1024 * 2);
    float* qkvf = (float*)alloc((size_t)M * 3072 * 4);
    bf16*  qbf  = (bf16*) alloc((size_t)M * 1024 * 2);
    bf16*  kbf  = (bf16*) alloc((size_t)M * 1024 * 2);
    bf16*  vTbf = (bf16*) alloc((size_t)M * 1024 * 2);
    bf16*  abf  = (bf16*) alloc((size_t)M * 1024 * 2);
    float* uf   = (float*)alloc((size_t)M * 5460 * 4);
    bf16*  gbf  = (bf16*) alloc((size_t)M * 2752 * 2);
    bf16*  wqkvT = (bf16*)alloc((size_t)3072 * 1024 * 2);
    bf16*  woT   = (bf16*)alloc((size_t)1024 * 1024 * 2);
    bf16*  w1T   = (bf16*)alloc((size_t)5504 * 1024 * 2);
    bf16*  w2T   = (bf16*)alloc((size_t)1024 * 2752 * 2);
    bf16*  logT  = (bf16*)alloc((size_t)32000 * 1024 * 2);

    auto gemm = [&](const bf16* A, const bf16* Bt, float* C, const float* bias,
                    const float* resid, int N, int Kp) {
        dim3 g((N + 63) / 64, M / 256);
        gemm_bf16_kernel<<<g, 256, 0, stream>>>(A, Bt, C, bias, resid, M, N, Kp);
    };
    auto wconv = [&](const float* W, bf16* Wt, int K, int N, int Kp, int Nrows) {
        long total = (long)Nrows * Kp;
        wconv_kernel<<<(total + 255) / 256, 256, 0, stream>>>(W, Wt, K, N, Kp, total);
    };

    embed_kernel<<<(M * 1024) / 256, 256, 0, stream>>>(tokens, token_emb, x);

    for (int l = 0; l < 4; ++l) {
        // --- attention ---
        wconv(wqkv + (size_t)l * 1024 * 3072, wqkvT, 1024, 3072, 1024, 3072);
        wconv(wo   + (size_t)l * 1024 * 1024, woT,   1024, 1024, 1024, 1024);
        rmsnorm_kernel<<<M, 256, 0, stream>>>(x, attn_norm_w + l * 1024, h);
        gemm(h, wqkvT, qkvf, nullptr, nullptr, 3072, 1024);
        rope_kernel<<<(32 * 2048 * 64) / 256, 256, 0, stream>>>(qkvf, qbf, kbf, vTbf);
        attn_kernel<<<dim3(128, 32), 32, 0, stream>>>(qbf, kbf, vTbf, abf);
        gemm(abf, woT, x2, nullptr, x, 1024, 1024);

        // --- GEGLU feedforward ---
        wconv(ff_w1 + (size_t)l * 1024 * 5460, w1T, 1024, 5460, 1024, 5504);
        wconv(ff_w2 + (size_t)l * 2730 * 1024, w2T, 2730, 1024, 2752, 1024);
        rmsnorm_kernel<<<M, 256, 0, stream>>>(x2, ff_norm_w + l * 1024, h);
        gemm(h, w1T, uf, ff_b1 + (size_t)l * 5460, nullptr, 5460, 1024);
        geglu_kernel<<<(M * 2752 + 255) / 256, 256, 0, stream>>>(uf, gbf);
        gemm(gbf, w2T, x, ff_b2 + (size_t)l * 1024, x2, 1024, 2752);
    }

    // --- final norm + logits ---
    rmsnorm_kernel<<<M, 256, 0, stream>>>(x, final_norm_w, h);
    wconv(logits_w, logT, 1024, 32000, 1024, 32000);
    gemm(h, logT, out, logits_b, nullptr, 32000, 1024);
}
